// TemporalAureliusGAT_83846351552519
// MI455X (gfx1250) — matrix operand order repaired and checked by reading, not compiled, not run on hardware
//
#include <hip/hip_runtime.h>
#include <hip/hip_bf16.h>
#include <math.h>

// ---------------- types for WMMA ----------------
typedef __attribute__((ext_vector_type(16))) __bf16 v16bf;
typedef __attribute__((ext_vector_type(8)))  __bf16 v8bf;
typedef __attribute__((ext_vector_type(8)))  float  v8f;

static constexpr int E   = 1536;   // embedding dim
static constexpr int D   = 128;    // hidden dim
static constexpr int TD  = 3 * D;  // 384 (gate width)
static constexpr int EPD = E + D;  // 1664 (classifier in-features)
static constexpr int KC  = 256;    // K-chunk staged in LDS per iteration
static constexpr int MT  = 32;     // rows per block (2 WMMA row-tiles)
static constexpr int APAD = 264;   // padded bf16 row stride (528 B) for LDS
static constexpr int HPAD = 132;   // padded f32 row stride for h tile

// pack 8 floats -> 8 bf16 (backend picks v_cvt_pk_bf16_f32 if available)
__device__ __forceinline__ v8bf pack8(const float4 a, const float4 b) {
  v8bf r;
  r[0] = (__bf16)a.x; r[1] = (__bf16)a.y; r[2] = (__bf16)a.z; r[3] = (__bf16)a.w;
  r[4] = (__bf16)b.x; r[5] = (__bf16)b.y; r[6] = (__bf16)b.z; r[7] = (__bf16)b.w;
  return r;
}

// B-fragment (32x16 bf16): lane half selects K half; 16 contiguous bf16.
__device__ __forceinline__ v16bf load_b_frag(const unsigned short* p) {
  v8bf lo = *reinterpret_cast<const v8bf*>(p);
  v8bf hi = *reinterpret_cast<const v8bf*>(p + 8);
  return __builtin_shufflevector(lo, hi, 0,1,2,3,4,5,6,7,8,9,10,11,12,13,14,15);
}

// A-fragment for row-tile from staged LDS buffer.
__device__ __forceinline__ v16bf load_a_frag(const unsigned short* p) {
  v8bf lo = *reinterpret_cast<const v8bf*>(p);
  v8bf hi = *reinterpret_cast<const v8bf*>(p + 8);
  return __builtin_shufflevector(lo, hi, 0,1,2,3,4,5,6,7,8,9,10,11,12,13,14,15);
}

// ------------- kernel 1: fp32 weights -> bf16 in workspace -------------
__global__ void cvt_weights_bf16(const float* __restrict__ wih,
                                 const float* __restrict__ whh,
                                 unsigned short* __restrict__ dst,
                                 int n_ih, int n_tot) {
  int i = blockIdx.x * blockDim.x + threadIdx.x;
  if (i < n_tot) {
    float f = (i < n_ih) ? wih[i] : whh[i - n_ih];
    __bf16 b = (__bf16)f;
    dst[i] = __builtin_bit_cast(unsigned short, b);
  }
}

// ------------- kernel 2: bulk copy memory -> updated_memory -------------
__global__ void copy_memory_f4(const float4* __restrict__ src,
                               float4* __restrict__ dst, int n4) {
  int i = blockIdx.x * blockDim.x + threadIdx.x;
  if (i < n4) dst[i] = src[i];
}

// ------------- kernel 3: fused GRU cell + scatter + classifier -------------
__global__ __launch_bounds__(256)
void gru_fused(const float* __restrict__ emb,
               const int*   __restrict__ nids,
               const float* __restrict__ mem,
               const float* __restrict__ b_ih,
               const float* __restrict__ b_hh,
               const float* __restrict__ cls_w,
               const float* __restrict__ cls_b,
               const unsigned short* __restrict__ wih_bf,
               const unsigned short* __restrict__ whh_bf,
               float* __restrict__ out_logits,
               float* __restrict__ out_mem,
               int N) {
  __shared__ __align__(16) unsigned short s_abuf[MT][APAD]; // 16.5 KB staged A
  __shared__ float s_h[MT][HPAD];                           // 16.5 KB new hidden
  __shared__ float s_red[8][MT][2];                         // 2 KB cls partials

  const int t     = threadIdx.x;
  const int wave  = t >> 5;
  const int lane  = t & 31;
  const int lrow  = lane & 15;
  const int khalf = lane >> 4;
  const int m0    = blockIdx.x * MT;

  // staging / classifier coords: thread owns (row=strow, part=spart)
  const int strow = t >> 3;   // 0..31
  const int spart = t & 7;    // 0..7  (8 column parts per row)
  int g_srow = m0 + strow; if (g_srow >= N) g_srow = N - 1;
  const float* emb_row = emb + (size_t)g_srow * E;

  float cls0 = 0.f, cls1 = 0.f;   // classifier partials, carried in registers

  const v8f vzero = {0.f,0.f,0.f,0.f,0.f,0.f,0.f,0.f};
  v8f accg[2][3];   // [row-tile][i_r / i_z / i_n]
  v8f acch[2][3];   // [row-tile][h_r / h_z / h_n]

  // wave w owns gate column tiles {w, w+8, w+16}: same 16 hidden cols in
  // each of the r/z/n blocks -> gates computed in registers, no transpose.
  const unsigned short* bih[3];
  const unsigned short* bhh[3];
  #pragma unroll
  for (int i = 0; i < 3; ++i) {
    const int n0 = (wave + 8 * i) * 16;
    bih[i] = wih_bf + (size_t)(n0 + lrow) * E + khalf * 16;
    bhh[i] = whh_bf + (size_t)(n0 + lrow) * D + khalf * 16;
    #pragma unroll
    for (int mt = 0; mt < 2; ++mt) { accg[mt][i] = vzero; acch[mt][i] = vzero; }
  }

  // ---------- phase 1: gi = emb @ w_ih^T, K = 1536 in 6 staged chunks ----------
  for (int kbase = 0; kbase < E; kbase += KC) {
    __syncthreads();   // previous chunk's fragment reads complete
    {                  // cooperative stage: 32 floats per thread
      const int c0 = spart * 32;
      const float* src = emb_row + kbase + c0;
      float4 f[8];
      #pragma unroll
      for (int q = 0; q < 8; ++q)
        f[q] = *reinterpret_cast<const float4*>(src + 4 * q);
      // classifier partial dot-products ride along for free
      const float* w0 = cls_w + kbase + c0;
      const float* w1 = w0 + EPD;
      #pragma unroll
      for (int q = 0; q < 8; ++q) {
        cls0 += f[q].x * w0[4*q+0] + f[q].y * w0[4*q+1]
              + f[q].z * w0[4*q+2] + f[q].w * w0[4*q+3];
        cls1 += f[q].x * w1[4*q+0] + f[q].y * w1[4*q+1]
              + f[q].z * w1[4*q+2] + f[q].w * w1[4*q+3];
      }
      #pragma unroll
      for (int q = 0; q < 4; ++q)
        *reinterpret_cast<v8bf*>(&s_abuf[strow][c0 + 8 * q]) =
            pack8(f[2 * q], f[2 * q + 1]);
    }
    __syncthreads();
    #pragma unroll
    for (int kk = 0; kk < KC / 32; ++kk) {
      const int c0 = kk * 32 + khalf * 16;
      const int k0 = kbase + kk * 32;
      // issue all loads of this K-step first: one wait, six WMMAs back-to-back
      v16bf a0 = load_a_frag(&s_abuf[lrow][c0]);
      v16bf a1 = load_a_frag(&s_abuf[16 + lrow][c0]);
      v16bf b0 = load_b_frag(bih[0] + k0);
      v16bf b1 = load_b_frag(bih[1] + k0);
      v16bf b2 = load_b_frag(bih[2] + k0);
      accg[0][0] = __builtin_amdgcn_wmma_f32_16x16x32_bf16(
          false, a0, false, b0, (short)0, accg[0][0], false, false);
      accg[1][0] = __builtin_amdgcn_wmma_f32_16x16x32_bf16(
          false, a1, false, b0, (short)0, accg[1][0], false, false);
      accg[0][1] = __builtin_amdgcn_wmma_f32_16x16x32_bf16(
          false, a0, false, b1, (short)0, accg[0][1], false, false);
      accg[1][1] = __builtin_amdgcn_wmma_f32_16x16x32_bf16(
          false, a1, false, b1, (short)0, accg[1][1], false, false);
      accg[0][2] = __builtin_amdgcn_wmma_f32_16x16x32_bf16(
          false, a0, false, b2, (short)0, accg[0][2], false, false);
      accg[1][2] = __builtin_amdgcn_wmma_f32_16x16x32_bf16(
          false, a1, false, b2, (short)0, accg[1][2], false, false);
    }
  }

  // ---------- phase 2: gh = prev @ w_hh^T, K = 128 (1 staged chunk) ----------
  __syncthreads();   // phase-1 fragment reads complete before restaging
  {                  // stage gathered prev rows: 16 floats per thread
    const int c0 = spart * 16;
    const size_t pn = (size_t)nids[g_srow];
    const float* src = mem + pn * D + c0;
    const float4 f0 = *reinterpret_cast<const float4*>(src);
    const float4 f1 = *reinterpret_cast<const float4*>(src + 4);
    const float4 f2 = *reinterpret_cast<const float4*>(src + 8);
    const float4 f3 = *reinterpret_cast<const float4*>(src + 12);
    *reinterpret_cast<v8bf*>(&s_abuf[strow][c0])     = pack8(f0, f1);
    *reinterpret_cast<v8bf*>(&s_abuf[strow][c0 + 8]) = pack8(f2, f3);
  }
  __syncthreads();
  #pragma unroll
  for (int kk = 0; kk < D / 32; ++kk) {
    const int c0 = kk * 32 + khalf * 16;
    const int k0 = kk * 32;
    v16bf a0 = load_a_frag(&s_abuf[lrow][c0]);
    v16bf a1 = load_a_frag(&s_abuf[16 + lrow][c0]);
    v16bf b0 = load_b_frag(bhh[0] + k0);
    v16bf b1 = load_b_frag(bhh[1] + k0);
    v16bf b2 = load_b_frag(bhh[2] + k0);
    acch[0][0] = __builtin_amdgcn_wmma_f32_16x16x32_bf16(
        false, a0, false, b0, (short)0, acch[0][0], false, false);
    acch[1][0] = __builtin_amdgcn_wmma_f32_16x16x32_bf16(
        false, a1, false, b0, (short)0, acch[1][0], false, false);
    acch[0][1] = __builtin_amdgcn_wmma_f32_16x16x32_bf16(
        false, a0, false, b1, (short)0, acch[0][1], false, false);
    acch[1][1] = __builtin_amdgcn_wmma_f32_16x16x32_bf16(
        false, a1, false, b1, (short)0, acch[1][1], false, false);
    acch[0][2] = __builtin_amdgcn_wmma_f32_16x16x32_bf16(
        false, a0, false, b2, (short)0, acch[0][2], false, false);
    acch[1][2] = __builtin_amdgcn_wmma_f32_16x16x32_bf16(
        false, a1, false, b2, (short)0, acch[1][2], false, false);
  }

  // ---------- phase 3: gates in registers + scatter updated memory ----------
  {
    const int d = wave * 16 + lrow;    // this lane's hidden column (all rows)
    const float bir = b_ih[d], biz = b_ih[D + d], bin = b_ih[2 * D + d];
    const float bhr = b_hh[d], bhz = b_hh[D + d], bhn = b_hh[2 * D + d];
    #pragma unroll
    for (int mt = 0; mt < 2; ++mt) {
      #pragma unroll
      for (int v = 0; v < 8; ++v) {
        const int mrow = mt * 16 + v + 8 * khalf;  // row within tile (D layout)
        const int grow = m0 + mrow;
        const int crow = grow < N ? grow : N - 1;
        const size_t gn = (size_t)nids[crow];
        const float pv = mem[gn * D + d];
        const float r = 1.f / (1.f + __expf(-(accg[mt][0][v] + bir + acch[mt][0][v] + bhr)));
        const float z = 1.f / (1.f + __expf(-(accg[mt][1][v] + biz + acch[mt][1][v] + bhz)));
        const float n = tanhf(accg[mt][2][v] + bin + r * (acch[mt][2][v] + bhn));
        const float h = (1.f - z) * n + z * pv;
        s_h[mrow][d] = h;
        if (grow < N) out_mem[gn * D + d] = h;
      }
    }
  }
  __syncthreads();

  // ---------- phase 4: classifier h-part + reduction ----------
  {
    const int d0 = spart * 16;
    const float* hv  = &s_h[strow][d0];
    const float* hw0 = cls_w + E + d0;
    const float* hw1 = cls_w + EPD + E + d0;
    #pragma unroll
    for (int j = 0; j < 16; ++j) {
      const float x = hv[j];
      cls0 += x * hw0[j];
      cls1 += x * hw1[j];
    }
    s_red[spart][strow][0] = cls0;
    s_red[spart][strow][1] = cls1;
  }
  __syncthreads();
  if (t < 64) {
    const int lm = t >> 1, c = t & 1;
    float s = cls_b[c];
    #pragma unroll
    for (int p = 0; p < 8; ++p) s += s_red[p][lm][c];
    const int grow = m0 + lm;
    if (grow < N) out_logits[(size_t)grow * 2 + c] = s;
  }
}

// --------------------------- launch ---------------------------
extern "C" void kernel_launch(void* const* d_in, const int* in_sizes, int n_in,
                              void* d_out, int out_size, void* d_ws, size_t ws_size,
                              hipStream_t stream) {
  const float* emb   = (const float*)d_in[0];
  const int*   nids  = (const int*)d_in[1];   // jnp.int64 -> int32 (x64 disabled)
  const float* mem   = (const float*)d_in[2];
  const float* w_ih  = (const float*)d_in[3];
  const float* w_hh  = (const float*)d_in[4];
  const float* b_ih  = (const float*)d_in[5];
  const float* b_hh  = (const float*)d_in[6];
  const float* cls_w = (const float*)d_in[7];
  const float* cls_b = (const float*)d_in[8];

  const int N         = in_sizes[1];          // 100000
  const int MAX_NODES = in_sizes[2] / D;      // 250000

  float* out_logits = (float*)d_out;
  float* out_mem    = (float*)d_out + (size_t)N * 2;

  unsigned short* wih_bf = (unsigned short*)d_ws;
  unsigned short* whh_bf = wih_bf + (size_t)TD * E;  // 589824 elems

  // 1) weights -> bf16 (1.25 MB in d_ws)
  {
    const int n_ih = TD * E, n_tot = TD * E + TD * D;
    cvt_weights_bf16<<<(n_tot + 255) / 256, 256, 0, stream>>>(
        w_ih, w_hh, wih_bf, n_ih, n_tot);
  }
  // 2) bulk copy persistent memory into updated_memory output
  {
    const int n4 = (MAX_NODES * D) / 4;
    copy_memory_f4<<<(n4 + 255) / 256, 256, 0, stream>>>(
        (const float4*)mem, (float4*)out_mem, n4);
  }
  // 3) fused GRU + scatter + classifier
  {
    const int grid = (N + MT - 1) / MT;
    gru_fused<<<grid, 256, 0, stream>>>(emb, nids, mem, b_ih, b_hh,
                                        cls_w, cls_b, wih_bf, whh_bf,
                                        out_logits, out_mem, N);
  }
}
